// QAT_ConvAttn_17901423690454
// MI455X (gfx1250) — 16-bit and 8-bit matrix operand order repaired and checked by reading, compiled, not run on hardware
//
#include <hip/hip_runtime.h>

#define NB_ 8
#define NC_ 64
#define NHC_ 32
#define NN_ 4096

typedef __attribute__((ext_vector_type(8)))  int      v8i;
typedef __attribute__((ext_vector_type(8)))  float    v8f;
typedef __attribute__((ext_vector_type(16))) _Float16 v16h;

union U8i  { v8i  v; unsigned u[8]; };
union U16h { v16h v; unsigned u[8]; };

#if defined(__has_builtin)
#if __has_builtin(__builtin_amdgcn_global_load_async_to_lds_b128) && \
    __has_builtin(__builtin_amdgcn_s_wait_asynccnt)
#define USE_ASYNC_LDS 1
#endif
#endif

#ifdef USE_ASYNC_LDS
typedef __attribute__((ext_vector_type(4))) int vi4;
typedef __attribute__((address_space(1))) vi4* gp_vi4;
typedef __attribute__((address_space(3))) vi4* lp_vi4;
__device__ __forceinline__ void async_copy_b128(const void* g, unsigned lds_off) {
  __builtin_amdgcn_global_load_async_to_lds_b128(
      (gp_vi4)(size_t)g, (lp_vi4)(size_t)lds_off, 0, 0);
}
#endif

__device__ __forceinline__ float group16_max(float x) {
  x = fmaxf(x, __shfl_xor(x, 1, 32));
  x = fmaxf(x, __shfl_xor(x, 2, 32));
  x = fmaxf(x, __shfl_xor(x, 4, 32));
  x = fmaxf(x, __shfl_xor(x, 8, 32));
  return x;
}
__device__ __forceinline__ float group16_sum(float x) {
  x += __shfl_xor(x, 1, 32);
  x += __shfl_xor(x, 2, 32);
  x += __shfl_xor(x, 4, 32);
  x += __shfl_xor(x, 8, 32);
  return x;
}

__global__ void __launch_bounds__(256)
wquant_kernel(const float* __restrict__ w, float* __restrict__ s_slot,
              signed char* __restrict__ w8, int n) {
  __shared__ float red[8];
  __shared__ float s_sh;
  int t = threadIdx.x;
  float m = 0.f;
  for (int i = t; i < n; i += 256) m = fmaxf(m, fabsf(w[i]));
  m = fmaxf(m, __shfl_xor(m, 1, 32));
  m = fmaxf(m, __shfl_xor(m, 2, 32));
  m = fmaxf(m, __shfl_xor(m, 4, 32));
  m = fmaxf(m, __shfl_xor(m, 8, 32));
  m = fmaxf(m, __shfl_xor(m, 16, 32));
  if ((t & 31) == 0) red[t >> 5] = m;
  __syncthreads();
  if (t == 0) {
    float mm = red[0];
    for (int i = 1; i < 8; ++i) mm = fmaxf(mm, red[i]);
    float s = mm * (1.0f / 127.0f);
    *s_slot = s;
    s_sh = s;
  }
  __syncthreads();
  float inv = 1.0f / s_sh;
  for (int i = t; i < n; i += 256) {
    float q = rintf(w[i] * inv);
    q = fminf(fmaxf(q, -127.f), 127.f);
    w8[i] = (signed char)q;
  }
}

__global__ void __launch_bounds__(256)
xquant_kernel(const float* __restrict__ x, const float* __restrict__ s_in,
              signed char* __restrict__ xq8) {
  int blk = blockIdx.x;
  int b  = blk >> 6;
  int nb = (blk & 63) * 64;
  int tx = threadIdx.x;
  int ty = threadIdx.y;
  float inv = 1.0f / s_in[0];
  int n = nb + tx;
  const float* xb = x + (size_t)b * NC_ * NN_;
  union { int4 i4; signed char c[16]; } tmp;
#pragma unroll
  for (int i = 0; i < 16; ++i) {
    int c = ty * 16 + i;
    float v = xb[(size_t)c * NN_ + n];
    float q = rintf(v * inv);
    q = fminf(fmaxf(q, -128.f), 127.f);
    tmp.c[i] = (signed char)q;
  }
  *(int4*)(xq8 + ((size_t)(b * NN_ + n)) * NC_ + ty * 16) = tmp.i4;
}

__global__ void __launch_bounds__(96)
qkv_kernel(const signed char* __restrict__ xq8,
           const signed char* __restrict__ w8base,
           const float* __restrict__ scales,
           const float* __restrict__ bq, const float* __restrict__ bk,
           const float* __restrict__ bv, const float* __restrict__ s_in_p,
           _Float16* __restrict__ Qf, _Float16* __restrict__ Kf,
           _Float16* __restrict__ VT) {
  int b  = blockIdx.x >> 8;
  int nb = (blockIdx.x & 255) * 16;
  int lane = threadIdx.x;
  int j    = threadIdx.y;
  int m   = lane & 15;
  int hf  = lane >> 4;
  int col = lane & 15;

  U8i a;
  const signed char* xrow = xq8 + ((size_t)(b * NN_ + nb)) * NC_ + m * NC_ + hf * 16;
  *(unsigned long long*)&a.u[0] = *(const unsigned long long*)(xrow + 0);
  *(unsigned long long*)&a.u[2] = *(const unsigned long long*)(xrow + 8);
  *(unsigned long long*)&a.u[4] = *(const unsigned long long*)(xrow + 32);
  *(unsigned long long*)&a.u[6] = *(const unsigned long long*)(xrow + 40);

  const signed char* w8 = w8base + j * (NHC_ * NC_);
  const float* bias = (j == 0) ? bq : (j == 1) ? bk : bv;
  float sc = s_in_p[0] * scales[j];

#pragma unroll
  for (int hb = 0; hb < 2; ++hb) {
    int hcoff = hb * 16;
    U8i bm;
    const signed char* wrow = w8 + (hcoff + col) * NC_ + hf * 16;
    *(int4*)&bm.u[0] = *(const int4*)(wrow + 0);
    *(int4*)&bm.u[4] = *(const int4*)(wrow + 32);

    v8i cz = {0, 0, 0, 0, 0, 0, 0, 0};
    v8i acc = __builtin_amdgcn_wmma_i32_16x16x64_iu8(
        true, a.v, true, bm.v, cz, false, false);
    float bsv = bias[hcoff + col];

    _Float16* dst;
    int stride;
    if (j == 2) {
      dst = VT + (size_t)b * NHC_ * NN_ + (size_t)(hcoff + col) * NN_ + nb;
      stride = 1;
    } else {
      _Float16* base = (j == 0) ? Qf : Kf;
      dst = base + ((size_t)(b * NN_ + nb)) * NHC_ + hcoff + col;
      stride = NHC_;
    }
#pragma unroll
    for (int r = 0; r < 8; ++r) {
      float val = (float)acc[r] * sc + bsv;
      dst[(r + 8 * hf) * stride] = (_Float16)val;
    }
  }
}

__global__ void __launch_bounds__(32)
attn_kernel(const _Float16* __restrict__ Qf, const _Float16* __restrict__ Kf,
            const _Float16* __restrict__ VT, const float* __restrict__ s_in_p,
            signed char* __restrict__ zq8) {
  __shared__ __align__(16) _Float16 Pl[16 * 32];
#ifdef USE_ASYNC_LDS
  __shared__ __align__(16) signed char Kls[2][2048];
  __shared__ __align__(16) signed char Vls[2][2048];
#endif
  int b  = blockIdx.x >> 8;
  int nb = (blockIdx.x & 255) * 16;
  int lane = threadIdx.x;
  int m   = lane & 15;
  int hf  = lane >> 4;
  int col = lane & 15;

  const _Float16* Qb = Qf + (size_t)b * NN_ * NHC_;
  const _Float16* Kb = Kf + (size_t)b * NN_ * NHC_;
  const _Float16* Vb = VT + (size_t)b * NHC_ * NN_;

  U16h aq;
  {
    const char* qrow = (const char*)Qb + (size_t)(nb + m) * 64 + hf * 32;
    *(int4*)&aq.u[0] = *(const int4*)(qrow + 0);
    *(int4*)&aq.u[4] = *(const int4*)(qrow + 16);
  }

  float mrow[8], lrow[8];
  v8f o0 = {0.f, 0.f, 0.f, 0.f, 0.f, 0.f, 0.f, 0.f};
  v8f o1 = {0.f, 0.f, 0.f, 0.f, 0.f, 0.f, 0.f, 0.f};
#pragma unroll
  for (int r = 0; r < 8; ++r) { mrow[r] = -3.0e38f; lrow[r] = 0.f; }

#ifdef USE_ASYNC_LDS
  auto issue_chunk = [&](int ck, int buf) {
    const char* kg = (const char*)Kb + (size_t)ck * 64;
    unsigned kl = (unsigned)(size_t)&Kls[buf][0];
#pragma unroll
    for (int i = 0; i < 4; ++i)
      async_copy_b128(kg + i * 512 + lane * 16, kl + i * 512 + lane * 16);
    int row = lane >> 2, part = lane & 3;
    const char* vg = (const char*)Vb + (size_t)ck * 2;
    unsigned vl = (unsigned)(size_t)&Vls[buf][0];
#pragma unroll
    for (int i = 0; i < 4; ++i) {
      int d = i * 8 + row;
      async_copy_b128(vg + (size_t)d * (NN_ * 2) + part * 16,
                      vl + d * 64 + part * 16);
    }
  };
  issue_chunk(0, 0);
  int buf = 0;
#endif

  for (int ck = 0; ck < NN_; ck += 32) {
    U16h bk0, bk1;
#ifdef USE_ASYNC_LDS
    if (ck + 32 < NN_) {
      issue_chunk(ck + 32, buf ^ 1);
      __builtin_amdgcn_s_wait_asynccnt(8);
    } else {
      __builtin_amdgcn_s_wait_asynccnt(0);
    }
    const char* kc = (const char*)&Kls[buf][0];
    *(int4*)&bk0.u[0] = *(const int4*)(kc + col * 64 + hf * 32);
    *(int4*)&bk0.u[4] = *(const int4*)(kc + col * 64 + hf * 32 + 16);
    *(int4*)&bk1.u[0] = *(const int4*)(kc + (16 + col) * 64 + hf * 32);
    *(int4*)&bk1.u[4] = *(const int4*)(kc + (16 + col) * 64 + hf * 32 + 16);
#else
    {
      const char* k0p = (const char*)Kb + (size_t)(ck + col) * 64 + hf * 32;
      const char* k1p = (const char*)Kb + (size_t)(ck + 16 + col) * 64 + hf * 32;
      *(int4*)&bk0.u[0] = *(const int4*)(k0p + 0);
      *(int4*)&bk0.u[4] = *(const int4*)(k0p + 16);
      *(int4*)&bk1.u[0] = *(const int4*)(k1p + 0);
      *(int4*)&bk1.u[4] = *(const int4*)(k1p + 16);
    }
#endif
    v8f czf = {0.f, 0.f, 0.f, 0.f, 0.f, 0.f, 0.f, 0.f};
    v8f s0 = __builtin_amdgcn_wmma_f32_16x16x32_f16(
        false, aq.v, false, bk0.v, (short)0, czf, false, false);
    v8f s1 = __builtin_amdgcn_wmma_f32_16x16x32_f16(
        false, aq.v, false, bk1.v, (short)0, czf, false, false);

#pragma unroll
    for (int r = 0; r < 8; ++r) {
      float x0 = s0[r], x1 = s1[r];
      float cm = group16_max(fmaxf(x0, x1));
      float mn = fmaxf(mrow[r], cm);
      float corr = __expf(mrow[r] - mn);
      float p0 = __expf(x0 - mn);
      float p1 = __expf(x1 - mn);
      float rs = group16_sum(p0 + p1);
      lrow[r] = lrow[r] * corr + rs;
      mrow[r] = mn;
      o0[r] *= corr;
      o1[r] *= corr;
      int row = r + 8 * hf;
      Pl[row * 32 + col]      = (_Float16)p0;
      Pl[row * 32 + 16 + col] = (_Float16)p1;
    }
    __syncthreads();

    U16h ap;
    {
      const char* pp = (const char*)Pl + m * 64 + hf * 32;
      *(int4*)&ap.u[0] = *(const int4*)(pp + 0);
      *(int4*)&ap.u[4] = *(const int4*)(pp + 16);
    }
    U16h bv0, bv1;
#ifdef USE_ASYNC_LDS
    const char* vc = (const char*)&Vls[buf][0];
    *(int4*)&bv0.u[0] = *(const int4*)(vc + col * 64 + hf * 32);
    *(int4*)&bv0.u[4] = *(const int4*)(vc + col * 64 + hf * 32 + 16);
    *(int4*)&bv1.u[0] = *(const int4*)(vc + (16 + col) * 64 + hf * 32);
    *(int4*)&bv1.u[4] = *(const int4*)(vc + (16 + col) * 64 + hf * 32 + 16);
    buf ^= 1;
#else
    {
      const char* v0p = (const char*)Vb + ((size_t)col * NN_ + ck) * 2 + hf * 32;
      const char* v1p = (const char*)Vb + ((size_t)(16 + col) * NN_ + ck) * 2 + hf * 32;
      *(int4*)&bv0.u[0] = *(const int4*)(v0p + 0);
      *(int4*)&bv0.u[4] = *(const int4*)(v0p + 16);
      *(int4*)&bv1.u[0] = *(const int4*)(v1p + 0);
      *(int4*)&bv1.u[4] = *(const int4*)(v1p + 16);
    }
    if (ck + 32 < NN_) {
      __builtin_prefetch((const char*)Kb + (size_t)(ck + 32 + col) * 64, 0, 3);
      __builtin_prefetch((const char*)Vb + ((size_t)col * NN_ + ck + 32) * 2, 0, 3);
    }
#endif
    o0 = __builtin_amdgcn_wmma_f32_16x16x32_f16(
        false, ap.v, false, bv0.v, (short)0, o0, false, false);
    o1 = __builtin_amdgcn_wmma_f32_16x16x32_f16(
        false, ap.v, false, bv1.v, (short)0, o1, false, false);
    __syncthreads();
  }

  float inv_s = 1.0f / s_in_p[0];
#pragma unroll
  for (int r = 0; r < 8; ++r) {
    float li = 1.0f / lrow[r];
    int tok = nb + r + 8 * hf;
    float q0 = fminf(fmaxf(rintf(o0[r] * li * inv_s), -128.f), 127.f);
    float q1 = fminf(fmaxf(rintf(o1[r] * li * inv_s), -128.f), 127.f);
    zq8[((size_t)(b * NN_ + tok)) * NHC_ + col]      = (signed char)q0;
    zq8[((size_t)(b * NN_ + tok)) * NHC_ + 16 + col] = (signed char)q1;
  }
}

__global__ void __launch_bounds__(128)
proj_out_kernel(const signed char* __restrict__ zq8,
                const signed char* __restrict__ wp8,
                const float* __restrict__ scales,
                const float* __restrict__ bp,
                const signed char* __restrict__ xq8,
                const float* __restrict__ s_in_p,
                const float* __restrict__ s_out_p,
                float* __restrict__ out) {
  int b  = blockIdx.x >> 8;
  int nb = (blockIdx.x & 255) * 16;
  int lane = threadIdx.x;
  int j    = threadIdx.y;
  int m   = lane & 15;
  int hf  = lane >> 4;
  int col = lane & 15;

  U8i a;
  {
    const signed char* zrow = zq8 + ((size_t)(b * NN_ + nb)) * NHC_ + m * NHC_ + hf * 16;
    *(unsigned long long*)&a.u[0] = *(const unsigned long long*)(zrow + 0);
    *(unsigned long long*)&a.u[2] = *(const unsigned long long*)(zrow + 8);
    a.u[4] = a.u[5] = a.u[6] = a.u[7] = 0u;
  }
  U8i bm;
  {
    const signed char* wrow = wp8 + (j * 16 + col) * NHC_ + hf * 16;
    *(int4*)&bm.u[0] = *(const int4*)wrow;
    bm.u[4] = bm.u[5] = bm.u[6] = bm.u[7] = 0u;
  }
  v8i cz = {0, 0, 0, 0, 0, 0, 0, 0};
  v8i acc = __builtin_amdgcn_wmma_i32_16x16x64_iu8(
      true, a.v, true, bm.v, cz, false, false);

  float s_in  = s_in_p[0];
  float s_out = s_out_p[0];
  float inv_so = 1.0f / s_out;
  float sc = s_in * scales[3];
  int c = j * 16 + col;
  float bsv = bp[c];
#pragma unroll
  for (int r = 0; r < 8; ++r) {
    int tok = nb + r + 8 * hf;
    float xq = (float)xq8[((size_t)(b * NN_ + tok)) * NC_ + c] * s_in;
    float o  = (float)acc[r] * sc + bsv + xq;
    float q  = fminf(fmaxf(rintf(o * inv_so), -128.f), 127.f);
    out[(size_t)b * NC_ * NN_ + (size_t)c * NN_ + tok] = q * s_out;
  }
}

extern "C" void kernel_launch(void* const* d_in, const int* in_sizes, int n_in,
                              void* d_out, int out_size, void* d_ws, size_t ws_size,
                              hipStream_t stream) {
  const float* x     = (const float*)d_in[0];
  const float* wq    = (const float*)d_in[1];
  const float* bq    = (const float*)d_in[2];
  const float* wk    = (const float*)d_in[3];
  const float* bk    = (const float*)d_in[4];
  const float* wv    = (const float*)d_in[5];
  const float* bv    = (const float*)d_in[6];
  const float* wp    = (const float*)d_in[7];
  const float* bp    = (const float*)d_in[8];
  const float* s_in  = (const float*)d_in[9];
  const float* s_out = (const float*)d_in[10];

  char* ws = (char*)d_ws;
  float*       scales = (float*)ws;
  signed char* wq8 = (signed char*)(ws + 256);
  signed char* wk8 = wq8 + NHC_ * NC_;
  signed char* wv8 = wk8 + NHC_ * NC_;
  signed char* wp8 = wv8 + NHC_ * NC_;
  signed char* xq8 = (signed char*)(ws + 16384);
  _Float16*    Qf  = (_Float16*)(ws + 16384 + (size_t)NB_ * NN_ * NC_);
  _Float16*    Kf  = Qf + (size_t)NB_ * NN_ * NHC_;
  _Float16*    VT  = Kf + (size_t)NB_ * NN_ * NHC_;
  signed char* zq8 = (signed char*)(VT + (size_t)NB_ * NN_ * NHC_);

  wquant_kernel<<<1, 256, 0, stream>>>(wq, scales + 0, wq8, NHC_ * NC_);
  wquant_kernel<<<1, 256, 0, stream>>>(wk, scales + 1, wk8, NHC_ * NC_);
  wquant_kernel<<<1, 256, 0, stream>>>(wv, scales + 2, wv8, NHC_ * NC_);
  wquant_kernel<<<1, 256, 0, stream>>>(wp, scales + 3, wp8, NC_ * NHC_);

  xquant_kernel<<<NB_ * (NN_ / 64), dim3(64, 4), 0, stream>>>(x, s_in, xq8);

  qkv_kernel<<<NB_ * (NN_ / 16), dim3(32, 3), 0, stream>>>(
      xq8, wq8, scales, bq, bk, bv, s_in, Qf, Kf, VT);

  attn_kernel<<<NB_ * (NN_ / 16), dim3(32, 1), 0, stream>>>(
      Qf, Kf, VT, s_in, zq8);

  proj_out_kernel<<<NB_ * (NN_ / 16), dim3(32, 4), 0, stream>>>(
      zq8, wp8, scales, bp, xq8, s_in, s_out, (float*)d_out);
}
